// GATClassifier_51539607725
// MI455X (gfx1250) — matrix operand order repaired and checked by reading, not compiled, not run on hardware
//
#include <hip/hip_runtime.h>
#include <hip/hip_bf16.h>

#define GAT_N 100000
#define GAT_H 4

typedef __attribute__((ext_vector_type(16))) __bf16 v16bf;
typedef __attribute__((ext_vector_type(8)))  float  v8f;

// ---------- order-preserving float <-> uint for atomicMax-based segment_max ----------
__device__ __forceinline__ unsigned f2ord(float f) {
  unsigned u = __float_as_uint(f);
  return (u & 0x80000000u) ? ~u : (u | 0x80000000u);
}
__device__ __forceinline__ float ord2f(unsigned u) {
  return (u & 0x80000000u) ? __uint_as_float(u & 0x7fffffffu) : __uint_as_float(~u);
}

// ---------- C = A(MxK,f32) * B(KxNCOL,f32) in f32 via bf16 WMMA ----------
// Block = 256 threads = 8 waves. Whole weight matrix B is staged transposed in
// LDS as bf16 (NCOL*K*2 bytes). Each wave computes a 16 x (16*TN) strip:
// A fragment loaded once per K-chunk, TN WMMAs issued against it.
// Fragment layouts as loaded here (wave32, contiguous k for A and B):
//   A 16x32 bf16 : lanes 0-15 row M=lane, elem j -> K=j;
//                  lanes 16-31 same rows, K offset +16.
//   B 32x16 bf16 : lanes 0-15 col N=lane, elem j -> K=j; lanes 16-31 -> K=16+j.
//   C/D 16x16 f32: vgpr r -> row r (lanes 0-15) / r+8 (lanes 16-31), col = lane&15.
template <int K, int NCOL, int TN>
__global__ void wmma_gemm_bf16(const float* __restrict__ A, const float* __restrict__ B,
                               float* __restrict__ C, int M) {
  __shared__ __bf16 Wt[NCOL * K];   // transposed: Wt[n*K + k]

  // cooperative transpose+convert of the weight matrix into LDS
  for (int idx = threadIdx.x; idx < NCOL * K; idx += blockDim.x) {
    int n = idx / K, k = idx % K;
    Wt[idx] = (__bf16)B[(size_t)k * NCOL + n];
  }
  __syncthreads();

  const int wid  = (int)((blockIdx.x * blockDim.x + threadIdx.x) >> 5);
  const int lane = (int)(threadIdx.x & 31u);
  const int wavesPerStrip = NCOL / (16 * TN);
  const int row0    = (wid / wavesPerStrip) << 4;
  const int colbase = (wid % wavesPerStrip) * (16 * TN);
  if (row0 >= M) return;

  const int r  = lane & 15;
  const int kg = lane >> 4;          // half-wave selector

  v8f acc[TN];
#pragma unroll
  for (int t = 0; t < TN; ++t) acc[t] = (v8f){};

#pragma unroll
  for (int kk = 0; kk < K; kk += 32) {
    v16bf a;
    const float* ap = A + (size_t)(row0 + r) * K + kk + kg * 16;
#pragma unroll
    for (int j = 0; j < 8; ++j) {
      a[j]     = (__bf16)ap[j];       // K = kk + kg*16 + j
      a[j + 8] = (__bf16)ap[8 + j];   // K = kk + kg*16 + 8 + j
    }
#pragma unroll
    for (int t = 0; t < TN; ++t) {
      v16bf b;
      const __bf16* bp = &Wt[(size_t)(colbase + t * 16 + r) * K + kk + kg * 16];
#pragma unroll
      for (int j = 0; j < 16; ++j) b[j] = bp[j];   // contiguous 32B from LDS
      acc[t] = __builtin_amdgcn_wmma_f32_16x16x32_bf16(false, a, false, b, (short)0,
                                                       acc[t], false, false);
    }
  }

#pragma unroll
  for (int t = 0; t < TN; ++t)
#pragma unroll
    for (int rr = 0; rr < 8; ++rr)
      C[(size_t)(row0 + kg * 8 + rr) * NCOL + colbase + t * 16 + r] = acc[t][rr];
}

// ---------- per-(node,head) attention logits: alpha = sum_c h[n,h,c]*a[h,c] ----------
__global__ void alpha_kernel(const float* __restrict__ h, const float* __restrict__ a_src,
                             const float* __restrict__ a_dst, float* __restrict__ as,
                             float* __restrict__ ad, int n, int ch) {
  int i = blockIdx.x * blockDim.x + threadIdx.x;
  if (i >= n * GAT_H) return;
  int node = i / GAT_H, hd = i % GAT_H;
  const float* hp = h + (size_t)node * GAT_H * ch + hd * ch;
  const float* sp = a_src + hd * ch;
  const float* dp = a_dst + hd * ch;
  float vs = 0.f, vd = 0.f;
  for (int c = 0; c < ch; ++c) { float v = hp[c]; vs += v * sp[c]; vd += v * dp[c]; }
  as[i] = vs; ad[i] = vd;
}

__device__ __forceinline__ void edge_ends(const int* __restrict__ ei, long i, int E,
                                          int n, int& s, int& d) {
  if (i < E) { s = ei[i]; d = ei[(size_t)E + i]; }
  else       { s = d = (int)(i - E); }            // appended self loops
}

// ---------- pass 1: segment_max of leaky_relu(as[src]+ad[dst]) over dst ----------
__global__ void edge_max_kernel(const int* __restrict__ ei, int E, int n,
                                const float* __restrict__ as, const float* __restrict__ ad,
                                unsigned* __restrict__ m) {
  long i = (long)blockIdx.x * blockDim.x + threadIdx.x;
  if (i >= (long)E + n) return;
  int s, d; edge_ends(ei, i, E, n, s, d);
#pragma unroll
  for (int h = 0; h < GAT_H; ++h) {
    float e = as[(size_t)s * GAT_H + h] + ad[(size_t)d * GAT_H + h];
    e = e > 0.f ? e : 0.2f * e;
    atomicMax(&m[(size_t)d * GAT_H + h], f2ord(e));
  }
}

// ---------- pass 2: segment_sum of exp(e - m[dst]) ----------
__global__ void edge_sum_kernel(const int* __restrict__ ei, int E, int n,
                                const float* __restrict__ as, const float* __restrict__ ad,
                                const unsigned* __restrict__ m, float* __restrict__ ssum) {
  long i = (long)blockIdx.x * blockDim.x + threadIdx.x;
  if (i >= (long)E + n) return;
  int s, d; edge_ends(ei, i, E, n, s, d);
#pragma unroll
  for (int h = 0; h < GAT_H; ++h) {
    float e = as[(size_t)s * GAT_H + h] + ad[(size_t)d * GAT_H + h];
    e = e > 0.f ? e : 0.2f * e;
    float mx = ord2f(m[(size_t)d * GAT_H + h]);
    atomicAdd(&ssum[(size_t)d * GAT_H + h], __expf(e - mx));
  }
}

// ---------- pass 3: out[dst] += alpha * h[src]; 32 lanes per edge ----------
__global__ void edge_agg_kernel(const int* __restrict__ ei, int E, int n,
                                const float* __restrict__ as, const float* __restrict__ ad,
                                const unsigned* __restrict__ m, const float* __restrict__ ssum,
                                const float* __restrict__ h, float* __restrict__ out, int ch) {
  long idx = (long)blockIdx.x * blockDim.x + threadIdx.x;
  int cht = GAT_H * ch;
  int cpl = cht >> 5;                 // channels per lane (4 for C=32, 2 for C=16)
  if (idx >= ((long)E + n) * 32) return;
  long i  = idx >> 5;
  int lane = (int)(idx & 31);
  int s, d; edge_ends(ei, i, E, n, s, d);
  int c0 = lane * cpl;
  int hd = c0 / ch;
  float e = as[(size_t)s * GAT_H + hd] + ad[(size_t)d * GAT_H + hd];
  e = e > 0.f ? e : 0.2f * e;
  float mx = ord2f(m[(size_t)d * GAT_H + hd]);
  float alpha = __expf(e - mx) / ssum[(size_t)d * GAT_H + hd];
  const float* hp = h + (size_t)s * cht + c0;
  float* op = out + (size_t)d * cht + c0;
  for (int c = 0; c < cpl; ++c) atomicAdd(&op[c], alpha * hp[c]);
}

// ---------- elementwise bias (+ optional ReLU) ----------
__global__ void bias_act_kernel(float* __restrict__ x, const float* __restrict__ b,
                                long tot, int cht, int do_relu) {
  long i = (long)blockIdx.x * blockDim.x + threadIdx.x;
  if (i >= tot) return;
  float v = x[i] + b[i % cht];
  x[i] = do_relu ? fmaxf(v, 0.f) : v;
}

// ---------- global_add_pool: pooled[c] = sum_n x[n,c] ----------
__global__ void pool_kernel(const float* __restrict__ x, float* __restrict__ pooled,
                            int n, int cht) {
  int t = blockIdx.x * blockDim.x + threadIdx.x;
  int c = t % cht;
  int rows = (gridDim.x * blockDim.x) / cht;
  float acc = 0.f;
  for (int r = t / cht; r < n; r += rows) acc += x[(size_t)r * cht + c];
  atomicAdd(&pooled[c], acc);
}

// ---------- final FC: out[o] = sum_c pooled[c]*fc_w[c,o] + fc_b[o] ----------
__global__ void final_kernel(const float* __restrict__ pooled, const float* __restrict__ fc_w,
                             const float* __restrict__ fc_b, float* __restrict__ out, int cht) {
  int o = threadIdx.x;
  if (o < 2) {
    float acc = fc_b[o];
    for (int c = 0; c < cht; ++c) acc += pooled[c] * fc_w[c * 2 + o];
    out[o] = acc;
  }
}

extern "C" void kernel_launch(void* const* d_in, const int* in_sizes, int n_in,
                              void* d_out, int out_size, void* d_ws, size_t ws_size,
                              hipStream_t stream) {
  const float* x      = (const float*)d_in[0];
  const int*   ei     = (const int*)d_in[1];
  const float* W1     = (const float*)d_in[2];
  const float* a_src1 = (const float*)d_in[3];
  const float* a_dst1 = (const float*)d_in[4];
  const float* b1     = (const float*)d_in[5];
  const float* W2     = (const float*)d_in[6];
  const float* a_src2 = (const float*)d_in[7];
  const float* a_dst2 = (const float*)d_in[8];
  const float* b2     = (const float*)d_in[9];
  const float* fc_w   = (const float*)d_in[10];
  const float* fc_b   = (const float*)d_in[11];
  float* out = (float*)d_out;

  const int N = GAT_N;
  const int E = in_sizes[1] / 2;
  const int C1 = 32, C2 = 16;
  const int CHT1 = GAT_H * C1;   // 128
  const int CHT2 = GAT_H * C2;   // 64

  // ---- workspace layout (regions reused between layer 1 and layer 2) ----
  char* ws = (char*)d_ws;
  size_t off = 0;
  auto carve = [&](size_t bytes) { char* p = ws + off; off += (bytes + 255) & ~(size_t)255; return p; };
  float*    h_buf  = (float*)   carve((size_t)N * CHT1 * 4); // h1; later h2 in first half
  float*    o_buf  = (float*)   carve((size_t)N * CHT1 * 4); // out1; later out2 in first half
  float*    as_buf = (float*)   carve((size_t)N * GAT_H * 4);
  float*    ad_buf = (float*)   carve((size_t)N * GAT_H * 4);
  unsigned* m_buf  = (unsigned*)carve((size_t)N * GAT_H * 4);
  float*    s_buf  = (float*)   carve((size_t)N * GAT_H * 4);
  float*    pooled = (float*)   carve(256);
  (void)ws_size;

  const int BT = 256;            // 8 waves per block
  long items   = (long)E + N;
  int  eblocks = (int)((items + BT - 1) / BT);
  int  ablocks = (int)((items * 32 + BT - 1) / BT);
  int  nhblk   = (N * GAT_H + BT - 1) / BT;

  // ================= layer 1 =================
  {
    // 6250 row strips x 2 waves (each wave: 16 rows x 64 cols) = 12500 waves
    int waves  = (N / 16) * (CHT1 / 64);
    int blocks = (waves * 32 + BT - 1) / BT;
    wmma_gemm_bf16<128, 128, 4><<<blocks, BT, 0, stream>>>(x, W1, h_buf, N);
  }
  alpha_kernel<<<nhblk, BT, 0, stream>>>(h_buf, a_src1, a_dst1, as_buf, ad_buf, N, C1);
  hipMemsetAsync(m_buf, 0, (size_t)N * GAT_H * 4, stream);
  hipMemsetAsync(s_buf, 0, (size_t)N * GAT_H * 4, stream);
  hipMemsetAsync(o_buf, 0, (size_t)N * CHT1 * 4, stream);
  edge_max_kernel<<<eblocks, BT, 0, stream>>>(ei, E, N, as_buf, ad_buf, m_buf);
  edge_sum_kernel<<<eblocks, BT, 0, stream>>>(ei, E, N, as_buf, ad_buf, m_buf, s_buf);
  edge_agg_kernel<<<ablocks, BT, 0, stream>>>(ei, E, N, as_buf, ad_buf, m_buf, s_buf,
                                              h_buf, o_buf, C1);
  {
    long tot = (long)N * CHT1;
    bias_act_kernel<<<(int)((tot + BT - 1) / BT), BT, 0, stream>>>(o_buf, b1, tot, CHT1, 1);
  }

  // ================= layer 2 =================
  float* h2   = h_buf;   // reuse (h1 dead)
  float* out2 = o_buf;   // reuse first half of out1 region after gemm2 consumed out1
  {
    // 6250 row strips x 1 wave (16 rows x 64 cols)
    int waves  = (N / 16) * (CHT2 / 64);
    int blocks = (waves * 32 + BT - 1) / BT;
    wmma_gemm_bf16<128, 64, 4><<<blocks, BT, 0, stream>>>(o_buf, W2, h2, N);
  }
  alpha_kernel<<<nhblk, BT, 0, stream>>>(h2, a_src2, a_dst2, as_buf, ad_buf, N, C2);
  hipMemsetAsync(m_buf, 0, (size_t)N * GAT_H * 4, stream);
  hipMemsetAsync(s_buf, 0, (size_t)N * GAT_H * 4, stream);
  hipMemsetAsync(out2, 0, (size_t)N * CHT2 * 4, stream);
  edge_max_kernel<<<eblocks, BT, 0, stream>>>(ei, E, N, as_buf, ad_buf, m_buf);
  edge_sum_kernel<<<eblocks, BT, 0, stream>>>(ei, E, N, as_buf, ad_buf, m_buf, s_buf);
  edge_agg_kernel<<<ablocks, BT, 0, stream>>>(ei, E, N, as_buf, ad_buf, m_buf, s_buf,
                                              h2, out2, C2);
  {
    long tot = (long)N * CHT2;
    bias_act_kernel<<<(int)((tot + BT - 1) / BT), BT, 0, stream>>>(out2, b2, tot, CHT2, 0);
  }

  // ================= pool + FC =================
  hipMemsetAsync(pooled, 0, 256, stream);
  pool_kernel<<<256, BT, 0, stream>>>(out2, pooled, N, CHT2);
  final_kernel<<<1, 32, 0, stream>>>(pooled, fc_w, fc_b, out, CHT2);
}
